// WTConv2d_75453985457521
// MI455X (gfx1250) — matrix operand order repaired and checked by reading, not compiled, not run on hardware
//
#include <hip/hip_runtime.h>

// ---------------------------------------------------------------------------
// WTConv2d fused pipeline for MI455X (gfx1250, wave32, WMMA + TDM).
//
// Dominant cost: 1x1 projection GEMM  M=512, K=1536, N=131072  (206 GFLOP).
// Strategy: bf16 activations/weights + v_wmma_f32_16x16x32_bf16 (f32 accum).
//   - halves the feats intermediate traffic (403 MB vs 805 MB in f32)
//   - 8x fewer matrix ops than the f32 16x16x4 WMMA path
// B tiles (32 pixels x 32 K, 2 KB) are streamed into LDS by the Tensor Data
// Mover (tensor_load_to_lds, TENSORcnt), double-buffered so the DMA for step
// k+1 overlaps the WMMAs of step k; one workgroup barrier per K step.
//
// d_ws layout: [0, 1.5MB)   A-fragment-swizzled bf16 w_proj
//              [2MB, ~405MB) bf16 feats, pixel-major [p][k], k contiguous
// ---------------------------------------------------------------------------

typedef __bf16 bf16_t;
typedef __attribute__((ext_vector_type(16))) __bf16 v16bf;
typedef __attribute__((ext_vector_type(8)))  float  v8f;
typedef unsigned int __attribute__((ext_vector_type(4))) u32x4;
typedef int  __attribute__((ext_vector_type(8))) i32x8;
typedef int  __attribute__((ext_vector_type(4))) i32x4;

#define Bn   8
#define Cn   128
#define Hn   256
#define Wn   256
#define hH   128          // H/2
#define hW   128          // W/2
#define Kdim 1536         // 12C
#define Mdim 512          // 4C
#define NPIX (Bn * hH * hW)   // 131072
#define KB   (Kdim / 32)      // 48 K-blocks

#ifndef __has_builtin
#define __has_builtin(x) 0
#endif
#if __has_builtin(__builtin_amdgcn_tensor_load_to_lds) && \
    __has_builtin(__builtin_amdgcn_s_wait_tensorcnt)
#define USE_TDM 1
#else
#define USE_TDM 0
#endif

// ---------------------------------------------------------------------------
// Kernel 1: f32 w_proj[512][1536] -> bf16, pre-swizzled into a
// 16-bit A-matrix 16x32 fragment layout:
//   lane L holds row M = L%16; K set = (L<16 ? {0..15} : {16..31}), contiguous, as B reads it
// stored so each lane's 16 halfs (32 B) are contiguous:
//   aswz[(((mb*KB)+kb)*32 + lane)*16 + i]
// ---------------------------------------------------------------------------
__global__ void __launch_bounds__(256)
swizzle_wproj(const float* __restrict__ wp, bf16_t* __restrict__ aswz) {
    int idx  = blockIdx.x * 256 + threadIdx.x;      // 0 .. 512*1536-1
    int i    = idx & 15;
    int t    = idx >> 4;
    int lane = t & 31;
    int t2   = t >> 5;
    int kb   = t2 % KB;
    int mb   = t2 / KB;
    int m    = mb * 16 + (lane & 15);
    int K    = kb * 32 + ((lane < 16) ? 0 : 16) + i;
    aswz[idx] = (bf16_t)wp[m * Kdim + K];
}

// ---------------------------------------------------------------------------
// Kernel 2: fused DWT + depthwise convs (k=1,3,5), output bf16 feats in
// pixel-major layout feats[p*1536 + k],  k = ksel*512 + q*128 + c,
// p = b*16384 + i*128 + j  (half-res pixel).
// xd is never materialized: xd[b][q*128+c][i][j] = x[b][c][2i+qi][2j+qj].
// Reads coalesced along j; LDS transpose makes writes coalesced along k.
// ---------------------------------------------------------------------------
__global__ void __launch_bounds__(256)
dwconv_feats(const float* __restrict__ x,
             const float* __restrict__ w1,
             const float* __restrict__ w3,
             const float* __restrict__ w5,
             bf16_t* __restrict__ feats) {
    const int i   = blockIdx.x;          // half-res row
    const int q   = blockIdx.y;          // DWT quad
    const int b   = blockIdx.z;          // batch
    const int qi  = q >> 1, qj = q & 1;
    const int tid = threadIdx.x;

    __shared__ __align__(64) bf16_t ls[3][32][128];   // [ksel][j][c], 24 KB

    for (int j0 = 0; j0 < hW; j0 += 32) {
        // ---- compute phase: thread -> (c, jj), j fastest for coalescing ----
        for (int idx = tid; idx < 128 * 32; idx += 256) {
            const int c   = idx >> 5;
            const int jj  = idx & 31;
            const int j   = j0 + jj;
            const int ch4 = q * 128 + c;
            const float* xb = x + ((size_t)(b * Cn + c)) * Hn * Wn;

            float t[5][5];
            #pragma unroll
            for (int di = -2; di <= 2; ++di) {
                const int  ii  = i + di;
                const bool vi  = (unsigned)ii < (unsigned)hH;
                const int  row = 2 * ii + qi;
                #pragma unroll
                for (int dj = -2; dj <= 2; ++dj) {
                    const int  jn = j + dj;
                    const bool vj = (unsigned)jn < (unsigned)hW;
                    t[di + 2][dj + 2] =
                        (vi && vj) ? xb[row * Wn + 2 * jn + qj] : 0.0f;
                }
            }
            float s1 = w1[ch4] * t[2][2];
            float s3 = 0.0f;
            #pragma unroll
            for (int di = 0; di < 3; ++di)
                #pragma unroll
                for (int dj = 0; dj < 3; ++dj)
                    s3 += w3[ch4 * 9 + di * 3 + dj] * t[di + 1][dj + 1];
            float s5 = 0.0f;
            #pragma unroll
            for (int di = 0; di < 5; ++di)
                #pragma unroll
                for (int dj = 0; dj < 5; ++dj)
                    s5 += w5[ch4 * 25 + di * 5 + dj] * t[di][dj];

            ls[0][jj][c] = (bf16_t)s1;
            ls[1][jj][c] = (bf16_t)s3;
            ls[2][jj][c] = (bf16_t)s5;
        }
        __syncthreads();
        // ---- write phase: coalesced 32 B chunks along k ----
        for (int wi = tid; wi < 32 * 3 * 8; wi += 256) {
            const int    part = wi & 7;            // 16-half chunk within c
            const int    ks   = (wi >> 3) % 3;     // kernel-size group
            const int    jj   = wi / 24;           // pixel in subtile
            const size_t p    = (size_t)(b * hH + i) * hW + (j0 + jj);
            v16bf v = *(const v16bf*)&ls[ks][jj][part * 16];
            *(v16bf*)(feats + p * Kdim + ks * 512 + q * 128 + part * 16) = v;
        }
        __syncthreads();
    }
}

// ---------------------------------------------------------------------------
// TDM: issue one 2D tile load  (32 pixel-rows x 32 bf16, row stride 1536)
// into LDS at lds_addr.  D# built per CDNA5 ISA:
//   group0: count=1 | lds_addr | global_addr[56:0] | type=2
//   group1: data_size=1 (2B), tensor_dim0=1536, tensor_dim1=NPIX,
//           tile_dim0=32, tile_dim1=32, tensor_dim0_stride=1536
// ---------------------------------------------------------------------------
#if USE_TDM
__device__ __forceinline__ void tdm_load_tile_b(const bf16_t* gptr,
                                                unsigned int lds_addr) {
    const unsigned long long ga = (unsigned long long)(uintptr_t)gptr;
    u32x4 g0;
    g0.x = 0x1u;                                        // count=1, user D#
    g0.y = lds_addr;                                    // LDS byte address
    g0.z = (unsigned int)(ga & 0xffffffffull);          // global_addr[31:0]
    g0.w = (unsigned int)((ga >> 32) & 0x1ffffffull)    // global_addr[56:32]
         | (2u << 30);                                  // type = 2 ("image")
    i32x8 g1;
    g1[0] = (1 << 16);                 // wg_mask=0, data_size=1 (2 bytes)
    g1[1] = (Kdim & 0xffff) << 16;     // atomic_barrier=0 | tensor_dim0 lo
    g1[2] = (Kdim >> 16) | ((NPIX & 0xffff) << 16);   // dim0 hi | dim1 lo
    g1[3] = (NPIX >> 16) | (32 << 16); // dim1 hi | tile_dim0 = 32
    g1[4] = 32;                        // tile_dim1 = 32, tile_dim2 = 0
    g1[5] = Kdim;                      // tensor_dim0_stride[31:0] = 1536
    g1[6] = 0;                         // stride0 hi | dim1_stride lo
    g1[7] = 0;
    i32x4 z4 = {0, 0, 0, 0};
#if __clang_major__ >= 23
    i32x8 z8 = {0, 0, 0, 0, 0, 0, 0, 0};
    __builtin_amdgcn_tensor_load_to_lds(g0, g1, z4, z4, z8, 0);
#else
    __builtin_amdgcn_tensor_load_to_lds(g0, g1, z4, z4, 0);
#endif
}
#endif

// ---------------------------------------------------------------------------
// Kernel 3: GEMM  y[m,p] = sum_k Wproj[m,k]*feats[k,p]  fused with IDWT
// scatter.  Block = 8 wave32s, owns full M=512 x 32 pixels.
// wave w -> M rows [64w, 64w+64): 4 m-subtiles x 2 n-subtiles = 8 WMMAs
// per 32-wide K step (48 steps).  B tile double-buffered in LDS, filled by
// the Tensor Data Mover (wave 0 issues, TENSORcnt-waited, barrier-published);
// lane L reads its B fragment as 32 contiguous bytes: pixel = ns*16 + L%16,
// K offset (L<16 ? 0 : 16)  (same contiguous k order as the A swizzle).
// C/D layout: VGPR r -> M = base + r + (L<16?0:8), N = L%16.
// ---------------------------------------------------------------------------
__global__ void __launch_bounds__(256)
gemm_idwt(const bf16_t* __restrict__ aswz,
          const bf16_t* __restrict__ feats,
          float* __restrict__ out) {
    __shared__ __align__(64) bf16_t lb[2][32][32];   // double-buffered B tile

    const int tid    = threadIdx.x;
    const int lane   = tid & 31;
    const int wave   = tid >> 5;
    const int n_base = blockIdx.x * 32;

    v8f acc[4][2] = {};

    // lane-resident base of this wave's A fragments (swizzled buffer)
    const bf16_t* aw = aswz + ((size_t)(wave * 4) * KB * 32 + lane) * 16;
    const bf16_t* btile0 = feats + (size_t)n_base * Kdim;   // + kb*32 per step

#if USE_TDM
    const unsigned int lds0 = (unsigned int)(uintptr_t)(&lb[0][0][0]);
    if (wave == 0)
        tdm_load_tile_b(btile0, lds0);                      // prologue: kb=0
#else
    const int pp = tid >> 3;          // staging: pixel
    const int ch = tid & 7;           // staging: 8-byte chunk within 32 K
    *(uint2*)(&lb[0][pp][ch * 4]) =
        *(const uint2*)(btile0 + (size_t)pp * Kdim + ch * 4);
#endif

    for (int kb = 0; kb < KB; ++kb) {
        const int cur = kb & 1;
        const int nxt = cur ^ 1;
#if USE_TDM
        if (wave == 0)
            __builtin_amdgcn_s_wait_tensorcnt(0);           // tile kb arrived
#endif
        __syncthreads();   // publish tile kb; all done reading buffer `nxt`
#if USE_TDM
        if (wave == 0 && kb + 1 < KB)
            tdm_load_tile_b(btile0 + (kb + 1) * 32, lds0 + nxt * 2048);
#else
        if (kb + 1 < KB)
            *(uint2*)(&lb[nxt][pp][ch * 4]) =
                *(const uint2*)(btile0 + (size_t)pp * Kdim + (kb + 1) * 32
                                + ch * 4);
#endif
        // prefetch next K-step of A (L2-resident stream)
        if (kb + 1 < KB)
            __builtin_prefetch(aw + (size_t)(kb + 1) * 32 * 16, 0, 1);

        v16bf a[4], bf[2];
        #pragma unroll
        for (int ms = 0; ms < 4; ++ms)
            a[ms] = *(const v16bf*)(aw + (size_t)(ms * KB + kb) * 32 * 16);
        #pragma unroll
        for (int ns = 0; ns < 2; ++ns) {
            const int pix = ns * 16 + (lane & 15);
            bf[ns] = *(const v16bf*)(&lb[cur][pix][(lane < 16) ? 0 : 16]);
        }
        #pragma unroll
        for (int ms = 0; ms < 4; ++ms)
            #pragma unroll
            for (int ns = 0; ns < 2; ++ns)
                acc[ms][ns] = __builtin_amdgcn_wmma_f32_16x16x32_bf16(
                    false, a[ms], false, bf[ns],
                    (short)0, acc[ms][ns], false, false);
    }

    // ---- epilogue: IDWT scatter straight from C fragments ----
    #pragma unroll
    for (int ms = 0; ms < 4; ++ms) {
        #pragma unroll
        for (int ns = 0; ns < 2; ++ns) {
            const int p  = n_base + ns * 16 + (lane & 15);
            const int b  = p >> 14;
            const int ij = p & 16383;
            const int i  = ij >> 7;
            const int j  = ij & 127;
            #pragma unroll
            for (int r = 0; r < 8; ++r) {
                const int m = wave * 64 + ms * 16 + r + ((lane < 16) ? 0 : 8);
                const int q = m >> 7;          // quad
                const int c = m & 127;         // channel
                out[((size_t)(b * Cn + c) * Hn + (2 * i + (q >> 1))) * Wn
                    + 2 * j + (q & 1)] = acc[ms][ns][r];
            }
        }
    }
}

// ---------------------------------------------------------------------------
extern "C" void kernel_launch(void* const* d_in, const int* in_sizes, int n_in,
                              void* d_out, int out_size, void* d_ws, size_t ws_size,
                              hipStream_t stream) {
    const float* x     = (const float*)d_in[0];
    const float* w1    = (const float*)d_in[1];
    const float* w3    = (const float*)d_in[2];
    const float* w5    = (const float*)d_in[3];
    const float* wproj = (const float*)d_in[4];
    float*       out   = (float*)d_out;

    // workspace: [0,1.5MB) swizzled A ; [2MB, 2MB+403MB) bf16 feats
    bf16_t* aswz  = (bf16_t*)d_ws;
    bf16_t* feats = (bf16_t*)((char*)d_ws + (size_t)(2u << 20));

    // 1) weight convert + A-fragment swizzle (786432 elems)
    swizzle_wproj<<<(Mdim * Kdim) / 256, 256, 0, stream>>>(wproj, aswz);

    // 2) fused DWT + depthwise convs -> bf16 feats (pixel-major)
    dim3 gdw(hH, 4, Bn);
    dwconv_feats<<<gdw, 256, 0, stream>>>(x, w1, w3, w5, feats);

    // 3) WMMA GEMM (TDM-staged B tiles) + fused IDWT scatter
    gemm_idwt<<<NPIX / 32, 256, 0, stream>>>(aswz, feats, out);
}
